// MultiHeadAttention_49211735277730
// MI455X (gfx1250) — matrix operand order repaired and checked by reading, not compiled, not run on hardware
//
#include <hip/hip_runtime.h>
#include <hip/hip_bf16.h>
#include <stdint.h>

// MI455X / gfx1250 multi-head attention.
// v_wmma_f32_16x16x32_f16 everywhere + GLOBAL_LOAD_ASYNC_TO_LDS_B128 double
// buffering (ASYNCcnt) for all LDS tile staging.
// Pipeline: prep (cvt x->f16, transpose W->f16) -> qkv GEMM -> flash attention
//           -> out GEMM.
// Workspace (~42MB): Q,K f16 [B,H,T,Dh]; V f16 [B,H,Dh,T]; Y f16 [B,T,C];
//                    Xh f16 [B*T,C]; WtA f16 [3C][C]; WtO f16 [C][C].

typedef __attribute__((ext_vector_type(16))) _Float16 v16h;
typedef __attribute__((ext_vector_type(8)))  _Float16 v8h;
typedef __attribute__((ext_vector_type(4)))  _Float16 v4h;
typedef __attribute__((ext_vector_type(8)))  float    v8f;

union AF { v16h v; v8h h[2]; };

#define NE   512
#define N3   1536
#define TT   2048
#define DH   64
#define NH   8

__device__ __forceinline__ v8f wmma_f16(v16h a, v16h b, v8f c) {
  // (neg_a, A, neg_b, B, c_mod, C, reuse_a, reuse_b)
  return __builtin_amdgcn_wmma_f32_16x16x32_f16(false, a, false, b, (short)0, c,
                                                false, false);
}

// Async DMA global->LDS, 16B per lane, tracked by ASYNCcnt (GVS addressing:
// saddr64 + vaddr32 + ioffset). vdst VGPR carries the per-lane LDS byte addr.
__device__ __forceinline__ void async_ld_b128(uint32_t lds_addr, uint32_t voff,
                                              const void* sbase) {
  asm volatile("global_load_async_to_lds_b128 %0, %1, %2"
               :: "v"(lds_addr), "v"(voff), "s"(sbase) : "memory");
}
__device__ __forceinline__ void wait_async0() {
  asm volatile("s_wait_asynccnt 0x0" ::: "memory");
}

// ---------------------------------------------------------------------------
// Prep: x f32 -> f16 (row-major), W f32 [K][N] -> f16 transposed [N][K].
// ---------------------------------------------------------------------------
__global__ __launch_bounds__(256) void cvt_x_kernel(const float* __restrict__ x,
                                                    _Float16* __restrict__ o) {
  const int i = (blockIdx.x * 256 + threadIdx.x) * 4;
  const float4 v = *(const float4*)(x + i);
  v4h h = { (_Float16)v.x, (_Float16)v.y, (_Float16)v.z, (_Float16)v.w };
  *(v4h*)(o + i) = h;
}

__global__ __launch_bounds__(256) void cvt_wt_kernel(const float* __restrict__ W,
                                                     _Float16* __restrict__ Wt,
                                                     int K, int N) {
  const int idx = blockIdx.x * 256 + threadIdx.x;
  const int k = idx / N, n = idx - k * N;
  Wt[(size_t)n * K + k] = (_Float16)W[idx];
}

// ---------------------------------------------------------------------------
// Kernel 1: qkv = Xh @ WtA^T + b_attn, scattered to Q/K (row) and V (transposed).
// 64x64 tile, 8 waves = 4 m-stripes x 2 n-halves, async double-buffered k-loop.
// ---------------------------------------------------------------------------
__global__ __launch_bounds__(256) void qkv_kernel(
    const _Float16* __restrict__ Xh, const _Float16* __restrict__ Wt,
    const float* __restrict__ bias,
    _Float16* __restrict__ Q, _Float16* __restrict__ K, _Float16* __restrict__ V)
{
  __shared__ _Float16 Xs[2][64][48];   // [m][k] tiles, row stride 96B
  __shared__ _Float16 Ws[2][64][48];   // [n][k] tiles

  const int tid  = threadIdx.x;
  const int wave = tid >> 5, lane = tid & 31;
  const int half = lane >> 4, lh = lane & 15;
  const int nb = blockIdx.x % (N3 / 64);
  const int mb = blockIdx.x / (N3 / 64);
  const int m0 = mb * 64, n0 = nb * 64;
  const int mw = (wave & 3) * 16;
  const int nw = (wave >> 2) * 32;

  // each thread DMAs one 16B chunk per tile: row r, byte seg s of a 64B row
  const int r = tid >> 2;
  const int s = (tid & 3) * 16;
  const uint32_t ldsrel = (uint32_t)(r * 96 + s);
  uint32_t xb[2], wb[2];
  xb[0] = (uint32_t)(uintptr_t)&Xs[0][0][0] + ldsrel;
  xb[1] = (uint32_t)(uintptr_t)&Xs[1][0][0] + ldsrel;
  wb[0] = (uint32_t)(uintptr_t)&Ws[0][0][0] + ldsrel;
  wb[1] = (uint32_t)(uintptr_t)&Ws[1][0][0] + ldsrel;
  const uint32_t voff = (uint32_t)(r * (NE * 2) + s);
  const _Float16* sx = Xh + (size_t)m0 * NE;
  const _Float16* sw = Wt + (size_t)n0 * NE;

  async_ld_b128(xb[0], voff, sx);
  async_ld_b128(wb[0], voff, sw);
  wait_async0();
  __syncthreads();

  v8f acc[2] = {};
  for (int kk = 0; kk < NE / 32; ++kk) {
    const int buf = kk & 1;
    if (kk + 1 < NE / 32) {
      async_ld_b128(xb[buf ^ 1], voff, sx + (kk + 1) * 32);
      async_ld_b128(wb[buf ^ 1], voff, sw + (kk + 1) * 32);
    }
    AF a;
    a.h[0] = *(const v8h*)&Xs[buf][mw + lh][half * 8];
    a.h[1] = *(const v8h*)&Xs[buf][mw + lh][16 + half * 8];
    #pragma unroll
    for (int j = 0; j < 2; ++j) {
      AF bf;
      bf.h[0] = *(const v8h*)&Ws[buf][nw + j * 16 + lh][half * 8];
      bf.h[1] = *(const v8h*)&Ws[buf][nw + j * 16 + lh][16 + half * 8];
      acc[j] = wmma_f16(a.v, bf.v, acc[j]);
    }
    wait_async0();
    __syncthreads();
  }

  const int mat  = n0 / NE;            // 0=Q 1=K 2=V
  const int head = (n0 % NE) / DH;
  #pragma unroll
  for (int j = 0; j < 2; ++j) {
    const int col = n0 + nw + j * 16 + lh;
    const float bv = bias[col];
    const int dh = col & (DH - 1);
    #pragma unroll
    for (int g = 0; g < 8; ++g) {
      const int mm = m0 + mw + g + half * 8;
      const int b = mm >> 11, t = mm & (TT - 1);
      const _Float16 hv = (_Float16)(acc[j][g] + bv);
      if (mat == 0)
        Q[((size_t)((b * NH + head) * TT + t)) * DH + dh] = hv;
      else if (mat == 1)
        K[((size_t)((b * NH + head) * TT + t)) * DH + dh] = hv;
      else  // V stored transposed [B,H,Dh,T] so attention DMAs it untouched
        V[((size_t)((b * NH + head) * DH + dh)) * TT + t] = hv;
    }
  }
}

// ---------------------------------------------------------------------------
// Kernel 2: flash attention. One block = (b,h) x 128 query rows; all 8 waves
// compute 16 rows each. K/V tiles (64 keys) async double-buffered in LDS.
// ---------------------------------------------------------------------------
__global__ __launch_bounds__(256) void attn_kernel(
    const _Float16* __restrict__ Qb, const _Float16* __restrict__ Kb,
    const _Float16* __restrict__ Vb, _Float16* __restrict__ Y)
{
  __shared__ _Float16 Ks[2][64][80];     // [tk][dh], row stride 160B
  __shared__ _Float16 Vt[2][64][80];     // [dh][tk] (already transposed in HBM)
  __shared__ _Float16 Ps[8][16][48];     // per-wave P restage (C->A layout)

  const int tid  = threadIdx.x;
  const int wave = tid >> 5, lane = tid & 31;
  const int half = lane >> 4, lh = lane & 15;
  const int qb = blockIdx.x & 15;        // T/128 = 16 query blocks
  const int bh = blockIdx.x >> 4;        // B*H  = 32
  const _Float16* Kplane = Kb + (size_t)bh * TT * DH;
  const _Float16* Vplane = Vb + (size_t)bh * DH * TT;
  const int q0 = qb * 128 + wave * 16;

  // async chunk geometry: 8KB tile = 512 x 16B chunks, 2 per thread
  const int c0 = tid, c1 = tid + 256;
  const int r0 = c0 >> 3, s0 = (c0 & 7) * 16;
  const int r1 = c1 >> 3, s1 = (c1 & 7) * 16;
  uint32_t kd[2][2], vd[2][2];
  #pragma unroll
  for (int b2 = 0; b2 < 2; ++b2) {
    kd[b2][0] = (uint32_t)(uintptr_t)&Ks[b2][0][0] + (uint32_t)(r0 * 160 + s0);
    kd[b2][1] = (uint32_t)(uintptr_t)&Ks[b2][0][0] + (uint32_t)(r1 * 160 + s1);
    vd[b2][0] = (uint32_t)(uintptr_t)&Vt[b2][0][0] + (uint32_t)(r0 * 160 + s0);
    vd[b2][1] = (uint32_t)(uintptr_t)&Vt[b2][0][0] + (uint32_t)(r1 * 160 + s1);
  }
  const uint32_t kv0 = (uint32_t)(r0 * (DH * 2) + s0);
  const uint32_t kv1 = (uint32_t)(r1 * (DH * 2) + s1);
  const uint32_t vv0 = (uint32_t)(r0 * (TT * 2) + s0);
  const uint32_t vv1 = (uint32_t)(r1 * (TT * 2) + s1);

  // Preload this wave's Q A-fragments (Dh=64 -> 2 k-chunks) from global.
  AF aq[2];
  {
    const _Float16* qrow = Qb + (size_t)bh * TT * DH + (size_t)(q0 + lh) * DH;
    #pragma unroll
    for (int kb = 0; kb < 2; ++kb) {
      aq[kb].h[0] = *(const v8h*)(qrow + kb * 32 + half * 8);
      aq[kb].h[1] = *(const v8h*)(qrow + kb * 32 + 16 + half * 8);
    }
  }

  float m[8], l[8];
  v8f o[4] = {};
  #pragma unroll
  for (int g = 0; g < 8; ++g) { m[g] = -1e30f; l[g] = 0.f; }

  // prologue DMA for tile 0
  async_ld_b128(kd[0][0], kv0, Kplane);
  async_ld_b128(kd[0][1], kv1, Kplane);
  async_ld_b128(vd[0][0], vv0, Vplane);
  async_ld_b128(vd[0][1], vv1, Vplane);
  wait_async0();
  __syncthreads();

  for (int it = 0; it < TT / 64; ++it) {
    const int buf = it & 1;
    if (it + 1 < TT / 64) {            // overlap next tile DMA with compute
      const int kt = (it + 1) * 64;
      async_ld_b128(kd[buf ^ 1][0], kv0, Kplane + (size_t)kt * DH);
      async_ld_b128(kd[buf ^ 1][1], kv1, Kplane + (size_t)kt * DH);
      async_ld_b128(vd[buf ^ 1][0], vv0, Vplane + kt);
      async_ld_b128(vd[buf ^ 1][1], vv1, Vplane + kt);
    }

    #pragma unroll
    for (int hc = 0; hc < 64; hc += 32) {      // 32 keys per softmax chunk
      // ---- S = Q K^T : two 16x16 tiles, 2 WMMA each over Dh ----
      v8f s[2] = {};
      #pragma unroll
      for (int n = 0; n < 2; ++n) {
        const int tkn = hc + n * 16;
        #pragma unroll
        for (int kb = 0; kb < 2; ++kb) {
          AF bf;
          bf.h[0] = *(const v8h*)&Ks[buf][tkn + lh][kb * 32 + half * 8];
          bf.h[1] = *(const v8h*)&Ks[buf][tkn + lh][kb * 32 + 16 + half * 8];
          s[n] = wmma_f16(aq[kb].v, bf.v, s[n]);
        }
      }
      // ---- online softmax (row lives per-VGPR, cols across 16 lanes) ----
      float alpha[8];
      #pragma unroll
      for (int g = 0; g < 8; ++g) {
        s[0][g] *= 0.125f; s[1][g] *= 0.125f;  // 1/sqrt(Dh)
        float mx = fmaxf(s[0][g], s[1][g]);
        mx = fmaxf(mx, __shfl_xor(mx, 1, 32));
        mx = fmaxf(mx, __shfl_xor(mx, 2, 32));
        mx = fmaxf(mx, __shfl_xor(mx, 4, 32));
        mx = fmaxf(mx, __shfl_xor(mx, 8, 32));
        const float mn = fmaxf(m[g], mx);
        alpha[g] = __expf(m[g] - mn);
        m[g] = mn;
        const float p0 = __expf(s[0][g] - mn);
        const float p1 = __expf(s[1][g] - mn);
        s[0][g] = p0; s[1][g] = p1;
        float rs = p0 + p1;
        rs += __shfl_xor(rs, 1, 32);
        rs += __shfl_xor(rs, 2, 32);
        rs += __shfl_xor(rs, 4, 32);
        rs += __shfl_xor(rs, 8, 32);
        l[g] = l[g] * alpha[g] + rs;
      }
      #pragma unroll
      for (int j = 0; j < 4; ++j)
        #pragma unroll
        for (int g = 0; g < 8; ++g) o[j][g] *= alpha[g];
      // ---- restage P (C-layout -> A-layout) through per-wave LDS ----
      #pragma unroll
      for (int g = 0; g < 8; ++g) {
        Ps[wave][g + half * 8][lh]      = (_Float16)s[0][g];
        Ps[wave][g + half * 8][16 + lh] = (_Float16)s[1][g];
      }
      AF ap;   // same-wave LDS RAW: DS ops issue in order per wave
      ap.h[0] = *(const v8h*)&Ps[wave][lh][half * 8];
      ap.h[1] = *(const v8h*)&Ps[wave][lh][16 + half * 8];
      // ---- O += P V : 4 WMMA over Dh column tiles ----
      #pragma unroll
      for (int j = 0; j < 4; ++j) {
        AF bv;
        bv.h[0] = *(const v8h*)&Vt[buf][j * 16 + lh][hc + half * 8];
        bv.h[1] = *(const v8h*)&Vt[buf][j * 16 + lh][hc + 16 + half * 8];
        o[j] = wmma_f16(ap.v, bv.v, o[j]);
      }
    }
    wait_async0();     // next tile landed (own wave); barrier covers the rest
    __syncthreads();
  }

  const int b = bh >> 3, h = bh & 7;
  #pragma unroll
  for (int j = 0; j < 4; ++j) {
    #pragma unroll
    for (int g = 0; g < 8; ++g) {
      const int t = q0 + g + half * 8;
      const float val = o[j][g] / l[g];
      Y[((size_t)(b * TT + t)) * NE + h * DH + j * 16 + lh] = (_Float16)val;
    }
  }
}

// ---------------------------------------------------------------------------
// Kernel 3: out = Y @ WtO^T + b_out (f32 output), async double-buffered.
// ---------------------------------------------------------------------------
__global__ __launch_bounds__(256) void proj_kernel(
    const _Float16* __restrict__ Yh, const _Float16* __restrict__ Wt,
    const float* __restrict__ bias, float* __restrict__ out)
{
  __shared__ _Float16 Xs[2][64][48];
  __shared__ _Float16 Ws[2][64][48];

  const int tid  = threadIdx.x;
  const int wave = tid >> 5, lane = tid & 31;
  const int half = lane >> 4, lh = lane & 15;
  const int nb = blockIdx.x & 7;           // 512/64
  const int mb = blockIdx.x >> 3;
  const int m0 = mb * 64, n0 = nb * 64;
  const int mw = (wave & 3) * 16;
  const int nw = (wave >> 2) * 32;

  const int r = tid >> 2;
  const int s = (tid & 3) * 16;
  const uint32_t ldsrel = (uint32_t)(r * 96 + s);
  uint32_t xb[2], wb[2];
  xb[0] = (uint32_t)(uintptr_t)&Xs[0][0][0] + ldsrel;
  xb[1] = (uint32_t)(uintptr_t)&Xs[1][0][0] + ldsrel;
  wb[0] = (uint32_t)(uintptr_t)&Ws[0][0][0] + ldsrel;
  wb[1] = (uint32_t)(uintptr_t)&Ws[1][0][0] + ldsrel;
  const uint32_t voff = (uint32_t)(r * (NE * 2) + s);
  const _Float16* sx = Yh + (size_t)m0 * NE;
  const _Float16* sw = Wt + (size_t)n0 * NE;

  async_ld_b128(xb[0], voff, sx);
  async_ld_b128(wb[0], voff, sw);
  wait_async0();
  __syncthreads();

  v8f acc[2] = {};
  for (int kk = 0; kk < NE / 32; ++kk) {
    const int buf = kk & 1;
    if (kk + 1 < NE / 32) {
      async_ld_b128(xb[buf ^ 1], voff, sx + (kk + 1) * 32);
      async_ld_b128(wb[buf ^ 1], voff, sw + (kk + 1) * 32);
    }
    AF a;
    a.h[0] = *(const v8h*)&Xs[buf][mw + lh][half * 8];
    a.h[1] = *(const v8h*)&Xs[buf][mw + lh][16 + half * 8];
    #pragma unroll
    for (int j = 0; j < 2; ++j) {
      AF bf;
      bf.h[0] = *(const v8h*)&Ws[buf][nw + j * 16 + lh][half * 8];
      bf.h[1] = *(const v8h*)&Ws[buf][nw + j * 16 + lh][16 + half * 8];
      acc[j] = wmma_f16(a.v, bf.v, acc[j]);
    }
    wait_async0();
    __syncthreads();
  }

  #pragma unroll
  for (int j = 0; j < 2; ++j) {
    const int col = n0 + nw + j * 16 + lh;
    const float bv = bias[col];
    #pragma unroll
    for (int g = 0; g < 8; ++g) {
      const int mrow = m0 + mw + g + half * 8;
      out[(size_t)mrow * NE + col] = acc[j][g] + bv;
    }
  }
}

// ---------------------------------------------------------------------------
extern "C" void kernel_launch(void* const* d_in, const int* in_sizes, int n_in,
                              void* d_out, int out_size, void* d_ws, size_t ws_size,
                              hipStream_t stream) {
  (void)in_sizes; (void)n_in; (void)out_size; (void)ws_size;
  const float* x      = (const float*)d_in[0];
  const float* W_attn = (const float*)d_in[1];
  const float* b_attn = (const float*)d_in[2];
  const float* W_out  = (const float*)d_in[3];
  const float* b_out  = (const float*)d_in[4];
  float* out = (float*)d_out;

  char* ws = (char*)d_ws;
  const size_t eight = (size_t)8 << 20;               // 8 MB per f16 tensor
  _Float16* Qb  = (_Float16*)(ws);
  _Float16* Kb  = (_Float16*)(ws + eight);
  _Float16* Vb  = (_Float16*)(ws + 2 * eight);        // transposed [B,H,Dh,T]
  _Float16* Yb  = (_Float16*)(ws + 3 * eight);
  _Float16* Xh  = (_Float16*)(ws + 4 * eight);
  _Float16* WtA = (_Float16*)(ws + 5 * eight);        // [1536][512] f16, 1.5MB
  _Float16* WtO = (_Float16*)(ws + 5 * eight + (size_t)N3 * NE * 2);  // 0.5MB

  cvt_x_kernel <<<dim3(4096), 256, 0, stream>>>(x, Xh);             // 4M elems
  cvt_wt_kernel<<<dim3(3072), 256, 0, stream>>>(W_attn, WtA, NE, N3);
  cvt_wt_kernel<<<dim3(1024), 256, 0, stream>>>(W_out,  WtO, NE, NE);
  // 8192/64 x 1536/64 = 3072 blocks
  qkv_kernel <<<dim3(3072), 256, 0, stream>>>(Xh, WtA, b_attn, Qb, Kb, Vb);
  // B*H * T/128 = 32 * 16 = 512 blocks, 8 compute waves each
  attn_kernel<<<dim3(512), 256, 0, stream>>>(Qb, Kb, Vb, Yb);
  // 8192/64 x 512/64 = 1024 blocks
  proj_kernel<<<dim3(1024), 256, 0, stream>>>(Yb, WtO, b_out, out);
}
